// Model_57071525430170
// MI455X (gfx1250) — compile-verified, matrix operand order checked by reading, not run on hardware
//
#include <hip/hip_runtime.h>
#include <hip/hip_bf16.h>

typedef __bf16 bf16;
typedef __attribute__((ext_vector_type(16))) __bf16 v16bf;
typedef __attribute__((ext_vector_type(8)))  __bf16 v8bf;
typedef __attribute__((ext_vector_type(4)))  __bf16 v4bf;
typedef __attribute__((ext_vector_type(8)))  float  v8f;

#define STR 40   // LDS row stride: 32 data + 8 pad bf16 (80 B/row, 16B-aligned)

// ---------------------------------------------------------------------------
// helpers
// ---------------------------------------------------------------------------

// CDNA5 async copy: 16 bytes global -> LDS, tracked by ASYNCcnt (no VGPR trip).
__device__ __forceinline__ void async_cp16(const bf16* g, bf16* l) {
  unsigned int lofs = (unsigned int)(size_t)l;  // low 32 bits = LDS offset
  asm volatile("global_load_async_to_lds_b128 %0, %1, off"
               :: "v"(lofs), "v"(g) : "memory");
}
__device__ __forceinline__ void wait_async0() {
  asm volatile("s_wait_asynccnt 0x0" ::: "memory");
}

// 16-element bf16 fragment: lane holds K = {h*8..h*8+7} ∪ {16+h*8..16+h*8+7}
__device__ __forceinline__ v16bf load_frag(const bf16* p) {
  v8bf lo = *(const v8bf*)(p);
  v8bf hi = *(const v8bf*)(p + 16);
  return __builtin_shufflevector(lo, hi, 0,1,2,3,4,5,6,7,8,9,10,11,12,13,14,15);
}

__device__ __forceinline__ v8f wmma_bf16(v16bf a, v16bf b, v8f c) {
  return __builtin_amdgcn_wmma_f32_16x16x32_bf16(false, a, false, b, (short)0, c,
                                                 false, false);
}

__device__ __forceinline__ float apply_act(float v, const float* bias, int gc, int act) {
  if (bias) v += bias[gc];
  if (act == 1)       v = fmaxf(v, 0.0f);
  else if (act == 2)  v = (v > 0.0f) ? v : 0.01f * v;   // leaky_relu
  return v;
}

// ---------------------------------------------------------------------------
// fp32 -> bf16 convert (plain) and convert+transpose (for B operands)
// ---------------------------------------------------------------------------
__global__ __launch_bounds__(256) void cvt_bf16_kernel(const float* __restrict__ src,
                                                       bf16* __restrict__ dst,
                                                       long count) {
  long i = (long)blockIdx.x * 256 + threadIdx.x;
  if (i < count) dst[i] = (bf16)src[i];
}

// src[K][N] -> dst[N][K]
__global__ __launch_bounds__(256) void cvt_t_bf16_kernel(const float* __restrict__ src,
                                                         bf16* __restrict__ dst,
                                                         int K, int N) {
  long i = (long)blockIdx.x * 256 + threadIdx.x;
  if (i >= (long)K * N) return;
  int n = (int)(i / K);
  int k = (int)(i - (long)n * K);
  dst[i] = (bf16)src[(long)k * N + n];
}

// ---------------------------------------------------------------------------
// row degree of A = (adj!=0) + I  ->  dinv = rsqrt(rowsum)
// ---------------------------------------------------------------------------
__global__ __launch_bounds__(256) void degree_rsqrt_kernel(const float* __restrict__ adj,
                                                           float* __restrict__ dinv,
                                                           int n) {
  int row = blockIdx.x;
  int tid = threadIdx.x;
  const float* arow = adj + (long)row * n;
  float s = 0.0f;
  for (int c = tid * 4; c < n; c += 256 * 4) {
    float4 v = *(const float4*)(arow + c);
    s += ((v.x != 0.0f) ? 1.0f : 0.0f) + ((v.y != 0.0f) ? 1.0f : 0.0f) +
         ((v.z != 0.0f) ? 1.0f : 0.0f) + ((v.w != 0.0f) ? 1.0f : 0.0f);
  }
  __shared__ float red[256];
  red[tid] = s;
  __syncthreads();
  for (int off = 128; off > 0; off >>= 1) {
    if (tid < off) red[tid] += red[tid + off];
    __syncthreads();
  }
  if (tid == 0) dinv[row] = rsqrtf(red[0] + 1.0f);  // +1 for eye()
}

// ---------------------------------------------------------------------------
// bf16 WMMA GEMM: C[M,N] = act(A[M,K] @ B[K,N] + bias), B given TRANSPOSED.
// Block tile 128x128, 8 waves (4 in M x 2 in N), wave tile 32x64 (8 WMMA).
// Double-buffered LDS, async global->LDS staging, software pipelined.
// Optional A row-gather with K split (feats = [y[i] ; y[6000+j]]).
// Requires N % 128 == 0, K % 32 == 0 (splitK % 32 == 0). OOB M rows clamped.
// ---------------------------------------------------------------------------
__global__ __launch_bounds__(256) void gemm_bf16_kernel(
    const bf16* __restrict__ A, int lda,
    const bf16* __restrict__ BT, int ldbt,          // B^T: [N][K]
    float* __restrict__ Cf, bf16* __restrict__ Cb,
    bf16* __restrict__ CbT, int ldc, int ldcT,
    const float* __restrict__ bias, int act,
    int M, int K,
    const int* __restrict__ g1, const int* __restrict__ g2,
    int gs, int splitK, int rowOff2) {
  __shared__ bf16 sA[2][128 * STR];
  __shared__ bf16 sB[2][128 * STR];

  const int tid   = threadIdx.x;
  const int lane  = tid & 31;
  const int wave  = tid >> 5;
  const int wm    = wave & 3;
  const int wn    = wave >> 2;
  const int mlane = lane & 15;
  const int half  = lane >> 4;
  const int blockM  = blockIdx.x * 128;
  const int colBase = blockIdx.y * 128;

  v8f acc[2][4];
#pragma unroll
  for (int i = 0; i < 2; ++i)
#pragma unroll
    for (int j = 0; j < 4; ++j)
#pragma unroll
      for (int e = 0; e < 8; ++e) acc[i][j][e] = 0.0f;

  // staging coords: 2 chunks of A + 2 chunks of B per thread per tile
  const int r0  = tid >> 2;            // 0..63
  const int r1  = r0 + 64;             // 64..127
  const int c8  = (tid & 3) * 8;       // 0,8,16,24

  auto stage = [&](int k0, int buf) {
    int acol = (g1 && k0 >= splitK) ? (k0 - splitK) : k0;
#pragma unroll
    for (int s = 0; s < 2; ++s) {
      int r = (s == 0) ? r0 : r1;
      int grow = blockM + r;
      if (grow >= M) grow = M - 1;     // clamp: OOB rows never stored
      long arow;
      if (g1) arow = (k0 < splitK) ? (long)g1[(long)grow * gs]
                                   : (long)(rowOff2 + g2[(long)grow * gs]);
      else    arow = grow;
      async_cp16(A + arow * (long)lda + acol + c8, &sA[buf][r * STR + c8]);
      async_cp16(BT + (long)(colBase + r) * ldbt + k0 + c8, &sB[buf][r * STR + c8]);
    }
  };

  const int ktiles = K >> 5;
  stage(0, 0);
  for (int kt = 0; kt < ktiles; ++kt) {
    wait_async0();          // my tile-kt copies have landed
    __syncthreads();        // everyone's have; everyone done reading other buf
    if (kt + 1 < ktiles) stage((kt + 1) * 32, (kt + 1) & 1);
    const int buf = kt & 1;

    const bf16* aB = &sA[buf][(wm * 32 + mlane) * STR + half * 8];
    v16bf a0 = load_frag(aB);
    v16bf a1 = load_frag(aB + 16 * STR);
    const bf16* bB = &sB[buf][(wn * 64 + mlane) * STR + half * 8];
    v16bf b0 = load_frag(bB);
    v16bf b1 = load_frag(bB + 16 * STR);
    v16bf b2 = load_frag(bB + 32 * STR);
    v16bf b3 = load_frag(bB + 48 * STR);

    acc[0][0] = wmma_bf16(a0, b0, acc[0][0]);
    acc[0][1] = wmma_bf16(a0, b1, acc[0][1]);
    acc[0][2] = wmma_bf16(a0, b2, acc[0][2]);
    acc[0][3] = wmma_bf16(a0, b3, acc[0][3]);
    acc[1][0] = wmma_bf16(a1, b0, acc[1][0]);
    acc[1][1] = wmma_bf16(a1, b1, acc[1][1]);
    acc[1][2] = wmma_bf16(a1, b2, acc[1][2]);
    acc[1][3] = wmma_bf16(a1, b3, acc[1][3]);
    __syncthreads();
  }

  // epilogue: C 16x16 layout -> VGPR v: M = v + half*8, N = lane%16
#pragma unroll
  for (int i = 0; i < 2; ++i) {
#pragma unroll
    for (int j = 0; j < 4; ++j) {
      int gc = colBase + wn * 64 + j * 16 + mlane;
#pragma unroll
      for (int v = 0; v < 8; ++v) {
        int gr = blockM + wm * 32 + i * 16 + half * 8 + v;
        if (gr < M) {
          float val = apply_act(acc[i][j][v], bias, gc, act);
          if (Cf)  Cf[(long)gr * ldc + gc] = val;
          if (Cb)  Cb[(long)gr * ldc + gc] = (bf16)val;
          if (CbT) CbT[(long)gc * ldcT + gr] = (bf16)val;
        }
      }
    }
  }
}

// ---------------------------------------------------------------------------
// SpMM: X = relu(A_hat @ Z + bias); A_hat built on the fly from adj + dinv:
// A_hat[i,j] = ((adj[i,j]!=0) + (i==j)) * dinv[i] * dinv[j].
// Z given transposed (ZT [N][n]) so staging is contiguous + async.
// Same 128x128 / 8-wave / 32x64 WMMA core, double-buffered.
// ---------------------------------------------------------------------------
__global__ __launch_bounds__(256) void spmm_ahat_kernel(
    const float* __restrict__ adj, const float* __restrict__ dinv,
    const bf16* __restrict__ ZT, int ldzt,
    const float* __restrict__ bias,
    float* __restrict__ Xf, bf16* __restrict__ Xb, int ldc,
    int n) {
  __shared__ bf16 sA[2][128 * STR];
  __shared__ bf16 sB[2][128 * STR];

  const int tid   = threadIdx.x;
  const int lane  = tid & 31;
  const int wave  = tid >> 5;
  const int wm    = wave & 3;
  const int wn    = wave >> 2;
  const int mlane = lane & 15;
  const int half  = lane >> 4;
  const int blockM  = blockIdx.x * 128;
  const int colBase = blockIdx.y * 128;

  v8f acc[2][4];
#pragma unroll
  for (int i = 0; i < 2; ++i)
#pragma unroll
    for (int j = 0; j < 4; ++j)
#pragma unroll
      for (int e = 0; e < 8; ++e) acc[i][j][e] = 0.0f;

  const int br0 = tid >> 2;
  const int bc8 = (tid & 3) * 8;

  auto stage = [&](int k0, int buf) {
    // B (ZT) rows: async contiguous
#pragma unroll
    for (int s = 0; s < 2; ++s) {
      int r = br0 + s * 64;
      async_cp16(ZT + (long)(colBase + r) * ldzt + k0 + bc8, &sB[buf][r * STR + bc8]);
    }
    // A_hat tile from adj: 128x32, 4 float4 per thread
    for (int it = tid; it < 1024; it += 256) {
      int r  = it >> 3;
      int c4 = it & 7;
      int grow = blockM + r;
      if (grow >= n) grow = n - 1;      // clamp: OOB rows never stored
      int gcol = k0 + c4 * 4;
      float4 av = *(const float4*)(adj + (long)grow * n + gcol);
      float4 dv = *(const float4*)(dinv + gcol);
      float di = dinv[grow];
      v4bf o;
      o[0] = (bf16)((((av.x != 0.0f) ? 1.0f : 0.0f) + ((grow == gcol + 0) ? 1.0f : 0.0f)) * di * dv.x);
      o[1] = (bf16)((((av.y != 0.0f) ? 1.0f : 0.0f) + ((grow == gcol + 1) ? 1.0f : 0.0f)) * di * dv.y);
      o[2] = (bf16)((((av.z != 0.0f) ? 1.0f : 0.0f) + ((grow == gcol + 2) ? 1.0f : 0.0f)) * di * dv.z);
      o[3] = (bf16)((((av.w != 0.0f) ? 1.0f : 0.0f) + ((grow == gcol + 3) ? 1.0f : 0.0f)) * di * dv.w);
      *(v4bf*)&sA[buf][r * STR + c4 * 4] = o;
    }
  };

  const int ktiles = n >> 5;
  stage(0, 0);
  for (int kt = 0; kt < ktiles; ++kt) {
    wait_async0();
    __syncthreads();
    if (kt + 1 < ktiles) stage((kt + 1) * 32, (kt + 1) & 1);
    const int buf = kt & 1;

    const bf16* aB = &sA[buf][(wm * 32 + mlane) * STR + half * 8];
    v16bf a0 = load_frag(aB);
    v16bf a1 = load_frag(aB + 16 * STR);
    const bf16* bB = &sB[buf][(wn * 64 + mlane) * STR + half * 8];
    v16bf b0 = load_frag(bB);
    v16bf b1 = load_frag(bB + 16 * STR);
    v16bf b2 = load_frag(bB + 32 * STR);
    v16bf b3 = load_frag(bB + 48 * STR);

    acc[0][0] = wmma_bf16(a0, b0, acc[0][0]);
    acc[0][1] = wmma_bf16(a0, b1, acc[0][1]);
    acc[0][2] = wmma_bf16(a0, b2, acc[0][2]);
    acc[0][3] = wmma_bf16(a0, b3, acc[0][3]);
    acc[1][0] = wmma_bf16(a1, b0, acc[1][0]);
    acc[1][1] = wmma_bf16(a1, b1, acc[1][1]);
    acc[1][2] = wmma_bf16(a1, b2, acc[1][2]);
    acc[1][3] = wmma_bf16(a1, b3, acc[1][3]);
    __syncthreads();
  }

#pragma unroll
  for (int i = 0; i < 2; ++i) {
#pragma unroll
    for (int j = 0; j < 4; ++j) {
      int gc = colBase + wn * 64 + j * 16 + mlane;
#pragma unroll
      for (int v = 0; v < 8; ++v) {
        int gr = blockM + wm * 32 + i * 16 + half * 8 + v;
        if (gr < n) {
          float val = fmaxf(acc[i][j][v] + bias[gc], 0.0f);  // relu
          Xf[(long)gr * ldc + gc] = val;
          Xb[(long)gr * ldc + gc] = (bf16)val;
        }
      }
    }
  }
}

// ---------------------------------------------------------------------------
// y[n,c,f] = relu(sum_l xs[n,l,f]*conv_w[c,l] + conv_b[c]) -> bf16 [n, 1536]
// ---------------------------------------------------------------------------
__global__ __launch_bounds__(256) void conv_comb_kernel(
    const float* __restrict__ x0, const float* __restrict__ x1,
    const float* __restrict__ x2,
    const float* __restrict__ conv_w, const float* __restrict__ conv_b,
    bf16* __restrict__ y, int total) {
  int idx = blockIdx.x * 256 + threadIdx.x;  // n*256 + f
  if (idx >= total) return;
  int nrow = idx >> 8;
  int f = idx & 255;
  float a = x0[idx], b = x1[idx], c = x2[idx];
#pragma unroll
  for (int cc = 0; cc < 6; ++cc) {
    float v = a * conv_w[cc * 3 + 0] + b * conv_w[cc * 3 + 1] +
              c * conv_w[cc * 3 + 2] + conv_b[cc];
    v = fmaxf(v, 0.0f);
    y[(long)nrow * 1536 + cc * 256 + f] = (bf16)v;
  }
}

// ---------------------------------------------------------------------------
// out[p] = sigmoid(dot(h3[p,:], w4))  — one wave per row (N=1 final layer)
// ---------------------------------------------------------------------------
__global__ __launch_bounds__(256) void mlp_out_kernel(const float* __restrict__ h3,
                                                      const float* __restrict__ w4,
                                                      float* __restrict__ out,
                                                      int P, int K) {
  int row  = blockIdx.x * 8 + (threadIdx.x >> 5);
  int lane = threadIdx.x & 31;
  if (row >= P) return;
  float s = 0.0f;
  for (int k = lane; k < K; k += 32) s += h3[(long)row * K + k] * w4[k];
#pragma unroll
  for (int off = 16; off > 0; off >>= 1) s += __shfl_xor(s, off, 32);
  if (lane == 0) out[row] = 1.0f / (1.0f + expf(-s));
}

// ---------------------------------------------------------------------------
// labels[p] = rel_matrix[i_p, j_p]
// ---------------------------------------------------------------------------
__global__ __launch_bounds__(256) void labels_kernel(const float* __restrict__ rel,
                                                     const int* __restrict__ ts,
                                                     float* __restrict__ out,
                                                     int P, int ndis) {
  int p = blockIdx.x * 256 + threadIdx.x;
  if (p < P) out[p] = rel[(long)ts[2 * p] * ndis + ts[2 * p + 1]];
}

// ---------------------------------------------------------------------------
// host launcher
// ---------------------------------------------------------------------------
extern "C" void kernel_launch(void* const* d_in, const int* in_sizes, int n_in,
                              void* d_out, int out_size, void* d_ws, size_t ws_size,
                              hipStream_t stream) {
  const int NC = 6000, ND = 2000, NN = 8000, FIN = 1024, F = 256, P = 16384;
  const int E = 3072, E2 = 1536, E4 = 768, E6 = 512;

  const float* adj    = (const float*)d_in[0];
  const float* circ   = (const float*)d_in[1];
  const float* disf   = (const float*)d_in[2];
  const float* rel    = (const float*)d_in[3];
  const int*   ts     = (const int*)d_in[4];
  const float* W_rna  = (const float*)d_in[5];
  const float* W_dis  = (const float*)d_in[6];
  const float* gcn_W  = (const float*)d_in[7];
  const float* gcn_b  = (const float*)d_in[8];
  const float* conv_w = (const float*)d_in[9];
  const float* conv_b = (const float*)d_in[10];
  const float* w1 = (const float*)d_in[11];
  const float* b1 = (const float*)d_in[12];
  const float* w2 = (const float*)d_in[13];
  const float* b2 = (const float*)d_in[14];
  const float* w3 = (const float*)d_in[15];
  const float* b3 = (const float*)d_in[16];
  const float* w4 = (const float*)d_in[17];
  float* out = (float*)d_out;

  // carve workspace
  char* wsp = (char*)d_ws;
  auto alloc = [&](size_t bytes) -> char* {
    char* p = wsp;
    wsp += (bytes + 255) & ~(size_t)255;
    return p;
  };
  bf16* featb = (bf16*)alloc((size_t)NN * FIN * 2);
  bf16* wrt   = (bf16*)alloc((size_t)F * FIN * 2);    // W_rna^T [256][1024]
  bf16* wdt   = (bf16*)alloc((size_t)F * FIN * 2);    // W_dis^T
  bf16* gwt   = (bf16*)alloc((size_t)2 * F * F * 2);  // gcn_W[l]^T
  bf16* w1t   = (bf16*)alloc((size_t)E2 * E * 2);     // w1^T [1536][3072]
  bf16* w2t   = (bf16*)alloc((size_t)E4 * E2 * 2);    // w2^T [768][1536]
  bf16* w3t   = (bf16*)alloc((size_t)E6 * E4 * 2);    // w3^T [512][768]
  float* dinv = (float*)alloc((size_t)NN * 4);
  float* x0f  = (float*)alloc((size_t)NN * F * 4);
  float* x1f  = (float*)alloc((size_t)NN * F * 4);
  float* x2f  = (float*)alloc((size_t)NN * F * 4);
  bf16* xb    = (bf16*)alloc((size_t)NN * F * 2);
  bf16* zbT   = (bf16*)alloc((size_t)F * NN * 2);     // z^T [256][8000]
  bf16* yb    = (bf16*)alloc((size_t)NN * 1536 * 2);
  bf16* h1b   = (bf16*)alloc((size_t)P * E2 * 2);
  bf16* h2b   = (bf16*)alloc((size_t)P * E4 * 2);
  float* h3f  = (float*)alloc((size_t)P * E6 * 4);
  (void)ws_size; (void)n_in; (void)in_sizes; (void)out_size;

  auto cvt = [&](const float* s, bf16* d, long cnt) {
    cvt_bf16_kernel<<<(unsigned)((cnt + 255) / 256), 256, 0, stream>>>(s, d, cnt);
  };
  auto cvtT = [&](const float* s, bf16* d, int K_, int N_) {
    long cnt = (long)K_ * N_;
    cvt_t_bf16_kernel<<<(unsigned)((cnt + 255) / 256), 256, 0, stream>>>(s, d, K_, N_);
  };

  // operand conversions (A operands plain, B operands transposed)
  cvt(circ,  featb,                 (long)NC * FIN);
  cvt(disf,  featb + (long)NC*FIN,  (long)ND * FIN);
  cvtT(W_rna, wrt, FIN, F);
  cvtT(W_dis, wdt, FIN, F);
  cvtT(gcn_W,             gwt,             F, F);
  cvtT(gcn_W + (long)F*F, gwt + (long)F*F, F, F);
  cvtT(w1, w1t, E,  E2);
  cvtT(w2, w2t, E2, E4);
  cvtT(w3, w3t, E4, E6);

  degree_rsqrt_kernel<<<NN, 256, 0, stream>>>(adj, dinv, NN);

  // x0 = [circ @ W_rna ; dis @ W_dis]  (fp32 for conv stack + bf16 for GCN)
  gemm_bf16_kernel<<<dim3((NC + 127) / 128, F / 128), 256, 0, stream>>>(
      featb, FIN, wrt, FIN, x0f, xb, nullptr, F, 0, nullptr, 0, NC, FIN,
      nullptr, nullptr, 0, 1 << 30, 0);
  gemm_bf16_kernel<<<dim3((ND + 127) / 128, F / 128), 256, 0, stream>>>(
      featb + (long)NC * FIN, FIN, wdt, FIN, x0f + (long)NC * F,
      xb + (long)NC * F, nullptr, F, 0, nullptr, 0, ND, FIN,
      nullptr, nullptr, 0, 1 << 30, 0);

  // 2 GCN layers: zT = (x @ W_l)^T ; x = relu(A_hat @ z + b_l)
  float* xlf[2] = {x1f, x2f};
  for (int l = 0; l < 2; ++l) {
    gemm_bf16_kernel<<<dim3((NN + 127) / 128, F / 128), 256, 0, stream>>>(
        xb, F, gwt + (long)l * F * F, F, nullptr, nullptr, zbT, F, NN,
        nullptr, 0, NN, F, nullptr, nullptr, 0, 1 << 30, 0);
    spmm_ahat_kernel<<<dim3((NN + 127) / 128, F / 128), 256, 0, stream>>>(
        adj, dinv, zbT, NN, gcn_b + (long)l * F, xlf[l], xb, F, NN);
  }

  // conv combine -> y bf16 [NN, 1536]
  conv_comb_kernel<<<(NN * F + 255) / 256, 256, 0, stream>>>(
      x0f, x1f, x2f, conv_w, conv_b, yb, NN * F);

  // MLP layer 1 with fused pair-gather: feats = [y[i] ; y[6000+j]]
  gemm_bf16_kernel<<<dim3(P / 128, E2 / 128), 256, 0, stream>>>(
      yb, 1536, w1t, E, nullptr, h1b, nullptr, E2, 0, b1, 2, P, E,
      ts, ts + 1, 2, 1536, NC);
  // MLP layer 2
  gemm_bf16_kernel<<<dim3(P / 128, E4 / 128), 256, 0, stream>>>(
      h1b, E2, w2t, E2, nullptr, h2b, nullptr, E4, 0, b2, 2, P, E2,
      nullptr, nullptr, 0, 1 << 30, 0);
  // MLP layer 3 (fp32 out for the final dot)
  gemm_bf16_kernel<<<dim3(P / 128, E6 / 128), 256, 0, stream>>>(
      h2b, E4, w3t, E4, h3f, nullptr, nullptr, E6, 0, b3, 2, P, E4,
      nullptr, nullptr, 0, 1 << 30, 0);

  // final sigmoid(h3 @ w4) and labels
  mlp_out_kernel<<<P / 8, 256, 0, stream>>>(h3f, w4, out, P, E6);
  labels_kernel<<<(P + 255) / 256, 256, 0, stream>>>(rel, ts, out + P, P, ND);
}
